// MTReadout_9096740733004
// MI455X (gfx1250) — matrix operand order repaired and checked by reading, not compiled, not run on hardware
//
#include <hip/hip_runtime.h>

typedef __attribute__((ext_vector_type(16))) _Float16 v16h;
typedef __attribute__((ext_vector_type(8)))  _Float16 v8h;
typedef __attribute__((ext_vector_type(4)))  _Float16 v4h;
typedef __attribute__((ext_vector_type(8)))  float    v8f;

#define Hdim 2048
#define Cdim 64
#define Nrows 65536
#define LN_EPS 1e-5f
#define ROWP 2056   // padded LDS row stride in halves (2048 + 8) -> conflict-free

// ---------------------------------------------------------------------------
// Kernel 0: fold gamma into W (-> f16 Wg), compute sWg[c] = sum_h Wg[c,h]
// and bc[c] = b[c] + sum_h beta[h]*W[c,h].  One block per output column c.
// ---------------------------------------------------------------------------
__global__ __launch_bounds__(256) void prep_kernel(
    const float* __restrict__ W, const float* __restrict__ gamma,
    const float* __restrict__ beta, const float* __restrict__ bias,
    _Float16* __restrict__ Wg, float* __restrict__ sWg, float* __restrict__ bc) {
  __shared__ float r0[256], r1[256];
  const int c = blockIdx.x;
  const int t = threadIdx.x;
  float swg = 0.f, bsum = 0.f;
  for (int h = t; h < Hdim; h += 256) {
    float w  = W[c * Hdim + h];
    float wg = w * gamma[h];
    Wg[c * Hdim + h] = (_Float16)wg;
    swg  += wg;
    bsum += beta[h] * w;
  }
  r0[t] = swg; r1[t] = bsum;
  __syncthreads();
  for (int s = 128; s > 0; s >>= 1) {
    if (t < s) { r0[t] += r0[t + s]; r1[t] += r1[t + s]; }
    __syncthreads();
  }
  if (t == 0) { sWg[c] = r0[0]; bc[c] = bias[c] + r1[0]; }
}

// ---------------------------------------------------------------------------
// Main kernel: one block = 16 rows of x, 4 waves = 4 x (16-col tiles of C=64).
// Phase 1: stream x tile once from HBM: per-row mean/rstd (shuffle reduce)
//          AND f32->f16 convert into padded LDS tile (shared A operand).
// Phase 2: K-loop of v_wmma_f32_16x16x32_f16 over H=2048 (64 steps),
//          A = f16 tile from LDS (2x ds_load_b128), B = Wg f16 (L2-resident).
// Phase 3: y = rstd*(acc - mu*sWg) + bc, softplus, store.
// ---------------------------------------------------------------------------
__global__ __launch_bounds__(128) void lnlin_kernel(
    const float* __restrict__ x, const _Float16* __restrict__ Wg,
    const float* __restrict__ sWg, const float* __restrict__ bc,
    float* __restrict__ out) {
  __shared__ float s_mu[16], s_rs[16];
  extern __shared__ _Float16 xs[];          // [16][ROWP] f16 tile, 65792 B
  const int tid  = threadIdx.x;
  const int wave = tid >> 5;
  const int lane = tid & 31;
  const int m0   = blockIdx.x * 16;

  // ---- Phase 1: stats + f16 staging (wave w handles rows w, w+4, w+8, w+12)
  for (int rr = 0; rr < 4; ++rr) {
    const int row = wave + rr * 4;
    const float4* xp = (const float4*)(x + (size_t)(m0 + row) * Hdim);
    _Float16* lrow = xs + row * ROWP;
    float s = 0.f, ss = 0.f;
#pragma unroll
    for (int j = 0; j < 16; ++j) {          // 32 lanes * 16 * float4 = 2048
      float4 v = xp[lane + j * 32];
      s  += v.x + v.y + v.z + v.w;
      ss += v.x * v.x + v.y * v.y + v.z * v.z + v.w * v.w;
      v4h hv;
      hv[0] = (_Float16)v.x; hv[1] = (_Float16)v.y;
      hv[2] = (_Float16)v.z; hv[3] = (_Float16)v.w;
      *(v4h*)(lrow + lane * 4 + j * 128) = hv;   // ds_store_b64, bank-clean
    }
#pragma unroll
    for (int d = 16; d >= 1; d >>= 1) {
      s  += __shfl_xor(s,  d, 32);
      ss += __shfl_xor(ss, d, 32);
    }
    if (lane == 0) {
      float mu  = s * (1.f / Hdim);
      float var = ss * (1.f / Hdim) - mu * mu;
      s_mu[row] = mu;
      s_rs[row] = rsqrtf(var + LN_EPS);
    }
  }
  __syncthreads();

  // ---- Phase 2: WMMA K-loop ----
  // A order used here (contiguous k): lane<16 -> K = k0..k0+15 (row=lane%16),
  //                          lane>=16 -> same rows, K = k0+16..k0+31.
  // B order used here (same k as A): lane%16 = column, lanes<16 hold K=k0..k0+15,
  //                          lanes>=16 hold K=k0+16..k0+31 (contiguous in Wg).
  const int hh   = lane >> 4;         // 0 or 1
  const int mr   = lane & 15;
  const int n0   = wave * 16;
  const int colg = n0 + mr;
  const _Float16* arow = xs + mr * ROWP + hh * 16;
  const _Float16* wcol = Wg + (size_t)colg * Hdim + hh * 16;

  v8f acc = {};
#pragma unroll 2
  for (int k0 = 0; k0 < Hdim; k0 += 32) {
    v8h alo = *(const v8h*)(arow + k0);        // ds_load_b128
    v8h ahi = *(const v8h*)(arow + k0 + 8);    // ds_load_b128
    v16h a  = __builtin_shufflevector(alo, ahi,
                0, 1, 2, 3, 4, 5, 6, 7, 8, 9, 10, 11, 12, 13, 14, 15);
    v16h bm = *(const v16h*)(wcol + k0);       // 32B global, L2-resident
    acc = __builtin_amdgcn_wmma_f32_16x16x32_f16(
        /*neg_a=*/false, a, /*neg_b=*/false, bm,
        /*c_mod=*/(short)0, acc, /*reuse_a=*/false, /*reuse_b=*/false);
  }

  // ---- Phase 3: epilogue ----
  // C/D layout: VGPR v -> M = v + (lane>=16 ? 8 : 0), N = lane%16.
  const float sw  = sWg[colg];
  const float bcv = bc[colg];
#pragma unroll
  for (int v = 0; v < 8; ++v) {
    const int r = v + hh * 8;
    float y  = s_rs[r] * (acc[v] - s_mu[r] * sw) + bcv;
    float sp = (y > 20.f) ? y : log1pf(expf(y));   // stable softplus
    out[(size_t)(m0 + r) * Cdim + colg] = sp;
  }
}

// ---------------------------------------------------------------------------
extern "C" void kernel_launch(void* const* d_in, const int* in_sizes, int n_in,
                              void* d_out, int out_size, void* d_ws, size_t ws_size,
                              hipStream_t stream) {
  const float* x     = (const float*)d_in[0];
  const float* gamma = (const float*)d_in[1];
  const float* beta  = (const float*)d_in[2];
  const float* W     = (const float*)d_in[3];
  const float* bias  = (const float*)d_in[4];
  float* out = (float*)d_out;

  // workspace layout: Wg (f16, C*H) | sWg (f32, C) | bc (f32, C)
  _Float16* Wg  = (_Float16*)d_ws;
  float*    sWg = (float*)((char*)d_ws + (size_t)Cdim * Hdim * sizeof(_Float16));
  float*    bc  = sWg + Cdim;

  prep_kernel<<<Cdim, 256, 0, stream>>>(W, gamma, beta, bias, Wg, sWg, bc);

  const size_t ldsBytes = (size_t)16 * ROWP * sizeof(_Float16);  // 65792 B
  lnlin_kernel<<<Nrows / 16, 128, ldsBytes, stream>>>(x, Wg, sWg, bc, out);
}
